// QINCo_80590766342848
// MI455X (gfx1250) — matrix operand order repaired and checked by reading, not compiled, not run on hardware
//
#include <hip/hip_runtime.h>
#include <hip/hip_bf16.h>

// ---------------------------------------------------------------------------
// QINCo candidate search + refinement for MI455X (gfx1250, wave32, WMMA, TDM)
// ---------------------------------------------------------------------------

typedef __bf16 bf16_t;
typedef __attribute__((ext_vector_type(16))) __bf16 v16bf;
typedef __attribute__((ext_vector_type(8)))  float  v8f;
typedef uint32_t u32x4 __attribute__((ext_vector_type(4)));
typedef uint32_t u32x8 __attribute__((ext_vector_type(8)));

#define N_E     65536
#define E_DIM   64
#define HIDDEN  256
#define NLAYERS 3
#define TOP_A   64
#define BATCH   4096
#define NPAIRS  (BATCH * TOP_A)

// ---------------- WMMA helpers (layouts per CDNA5 ISA 7.12.2) ---------------

__device__ __forceinline__ v8f wmma_bf16(v16bf a, v16bf b, v8f c) {
  return __builtin_amdgcn_wmma_f32_16x16x32_bf16(
      /*neg_a=*/false, a, /*neg_b=*/false, b,
      /*c_mod=*/(short)0, c, /*reuse_a=*/false, /*reuse_b=*/false);
}

// A: 16x32 bf16, row-major source with row stride `ld`.
// lane (m = lane&15, hi = lane>>4) holds K = {8*hi+0..7} U {16+8*hi+0..7}.
__device__ __forceinline__ v16bf load_a_frag(const bf16_t* src, int ld,
                                             int row0, int k0, int lane) {
  int m = lane & 15, hi = lane >> 4;
  const bf16_t* p0 = src + (size_t)(row0 + m) * ld + k0 + 8 * hi;
  const bf16_t* p1 = p0 + 16;
  v16bf a;
#pragma unroll
  for (int e = 0; e < 8; ++e) { a[e] = p0[e]; a[e + 8] = p1[e]; }
  return a;
}

// B: 32x16 bf16 = (K x N). Source is the TRANSPOSED weight Wt[n][k] with
// row stride `ldk`. lane (n = lane&15, hi = lane>>4) holds K = {8*hi+0..7} U {16+8*hi+0..7}.
__device__ __forceinline__ v16bf load_b_frag(const bf16_t* wt, int ldk,
                                             int n0, int k0, int lane) {
  int n = lane & 15, hi = lane >> 4;
  const bf16_t* p = wt + (size_t)(n0 + n) * ldk + k0 + 8 * hi;
  v16bf b;
#pragma unroll
  for (int e = 0; e < 8; ++e) { b[e] = p[e]; b[e + 8] = p[e + 16]; }
  return b;
}

// ---------------- Tensor Data Mover: contiguous 1-D tile -> LDS -------------
// 2-group D# (VADDR2/VADDR3 = NULL): n8 elements of 8 bytes, tile_dim0 = n8
// (n8 <= 16384 here, fits the 16-bit tile_dim0). Issued by one wave; the
// issuer drains TENSORcnt, then a workgroup barrier publishes the data.
__device__ __forceinline__ void tdm_load_to_lds(const void* gptr,
                                                uint32_t lds_off, uint32_t n8) {
  uint64_t ga = (uint64_t)(uintptr_t)gptr;
  u32x4 g0;
  g0[0] = 1u;                                   // count=1, user descriptor
  g0[1] = lds_off;                              // lds_addr (bytes)
  g0[2] = (uint32_t)ga;                         // global_addr[31:0]
  g0[3] = (uint32_t)((ga >> 32) & 0x01FFFFFFu)  // global_addr[56:32]
          | 0x80000000u;                        // type = 2 ("image")
  u32x8 g1;
  g1[0] = 3u << 16;                             // data_size = 8 bytes
  g1[1] = (n8 & 0xFFFFu) << 16;                 // tensor_dim0[15:0]
  g1[2] = (n8 >> 16) | (1u << 16);              // tensor_dim0[31:16] | tensor_dim1=1
  g1[3] = (n8 & 0xFFFFu) << 16;                 // tile_dim0 = n8
  g1[4] = 1u;                                   // tile_dim1 = 1
  g1[5] = n8;                                   // tensor_dim0_stride
  g1[6] = 0u;
  g1[7] = 0u;
  asm volatile("tensor_load_to_lds %0, %1" :: "s"(g0), "s"(g1) : "memory");
}

// ---------------- small conversion / prep kernels ---------------------------

__global__ void k_cvt(const float* __restrict__ src, bf16_t* __restrict__ dst, int n) {
  int i = blockIdx.x * 256 + threadIdx.x;
  if (i < n) dst[i] = (bf16_t)src[i];
}

// dst[n*K+k] = src[k*N+n]  (transpose to Wt[n][k] for the B-frag loader)
__global__ void k_cvt_t(const float* __restrict__ src, bf16_t* __restrict__ dst,
                        int K, int N) {
  int i = blockIdx.x * 256 + threadIdx.x;
  if (i < K * N) {
    int n = i / K, k = i - n * K;
    dst[i] = (bf16_t)src[(size_t)k * N + n];
  }
}

__global__ void k_sqnorm(const float* __restrict__ src, float* __restrict__ dst, int rows) {
  int r = blockIdx.x * 256 + threadIdx.x;
  if (r < rows) {
    const float* p = src + (size_t)r * E_DIM;
    float s = 0.f;
#pragma unroll 8
    for (int d = 0; d < E_DIM; ++d) s += p[d] * p[d];
    dst[r] = s;
  }
}

__global__ void k_zero(float* __restrict__ counts, float* __restrict__ lossAcc) {
  int i = blockIdx.x * 256 + threadIdx.x;
  if (i < N_E) counts[i] = 0.f;
  if (i == 0) *lossAcc = 0.f;
}

// ---------------- phase 1: WMMA distances + partition-min top-64 ------------
// grid = BATCH/16 blocks, block = 128 threads (4 waves).
// Each block: 16 residual rows vs all 65536 codes (K = 64 => 2 WMMAs/tile).
// lane keeps running minima for 8 rows of column-class (wave*16 + lane&15).
__global__ void k_dist_topk(const bf16_t* __restrict__ res_bf,
                            const bf16_t* __restrict__ cb_bf,
                            const float* __restrict__ r2,
                            const float* __restrict__ c2,
                            int* __restrict__ candIdx) {
  const int wave = threadIdx.x >> 5;
  const int lane = threadIdx.x & 31;
  const int m = lane & 15, hi = lane >> 4;
  const int rowbase = blockIdx.x * 16;

  // A frags are invariant over the whole codebook sweep.
  v16bf a0 = load_a_frag(res_bf, E_DIM, rowbase, 0, lane);
  v16bf a1 = load_a_frag(res_bf, E_DIM, rowbase, 32, lane);

  float r2v[8];
#pragma unroll
  for (int r = 0; r < 8; ++r) r2v[r] = r2[rowbase + 8 * hi + r];

  float best[8];
  int   bidx[8];
#pragma unroll
  for (int r = 0; r < 8; ++r) { best[r] = 3.4e38f; bidx[r] = 0; }

  for (int t = wave; t < N_E / 16; t += 4) {
    const int cbase = t * 16;
    if (t + 4 < N_E / 16)
      __builtin_prefetch(cb_bf + (size_t)(cbase + 64) * E_DIM, 0, 3);
    v16bf b0 = load_b_frag(cb_bf, E_DIM, cbase, 0, lane);
    v16bf b1 = load_b_frag(cb_bf, E_DIM, cbase, 32, lane);
    v8f acc = {};
    acc = wmma_bf16(a0, b0, acc);
    acc = wmma_bf16(a1, b1, acc);
    const int col = cbase + m;            // C/D layout: lane%16 = column
    const float c2v = c2[col];
#pragma unroll
    for (int r = 0; r < 8; ++r) {
      float dist = r2v[r] - 2.f * acc[r] + c2v;   // VGPR r = row (+8 for hi)
      if (dist < best[r]) { best[r] = dist; bidx[r] = col; }
    }
  }

  const int cls = wave * 16 + m;          // 64 column classes per row
#pragma unroll
  for (int r = 0; r < 8; ++r)
    candIdx[(size_t)(rowbase + 8 * hi + r) * TOP_A + cls] = bidx[r];
}

// ---------------- phase 2: batched MLP refinement (bf16 WMMA + TDM) ---------
// block = 256 threads (8 waves), 64 pairs/block, grid = NPAIRS/64 = 4096.
// LDS: sX 16K | sH 32K | sT 32K | sW 128K | sCI 256B | sDist 256B = 213504 B.

template <int KDIM, bool ADD_OLD>
__device__ __forceinline__ void mlp_gemm(const bf16_t* __restrict__ sA,
                                         const bf16_t* __restrict__ sW,
                                         const float* __restrict__ bias_g,
                                         bf16_t* __restrict__ sO,
                                         int wave, int lane) {
  const int mt  = wave & 3;           // row tile: rows mt*16 .. +15
  const int ntb = (wave >> 2) * 8;    // 8 column tiles of 16
  const int n = lane & 15, hi = lane >> 4;

  v8f acc[8];
#pragma unroll
  for (int t = 0; t < 8; ++t) {
    float bv = bias_g[(ntb + t) * 16 + n];
#pragma unroll
    for (int r = 0; r < 8; ++r) acc[t][r] = bv;
  }
#pragma unroll
  for (int ks = 0; ks < KDIM; ks += 32) {
    v16bf a = load_a_frag(sA, KDIM, mt * 16, ks, lane);
#pragma unroll
    for (int t = 0; t < 8; ++t) {
      v16bf b = load_b_frag(sW, KDIM, (ntb + t) * 16, ks, lane);
      acc[t] = wmma_bf16(a, b, acc[t]);
    }
  }
#pragma unroll
  for (int t = 0; t < 8; ++t) {
    int col = (ntb + t) * 16 + n;
#pragma unroll
    for (int r = 0; r < 8; ++r) {
      int row = mt * 16 + 8 * hi + r;
      float v = acc[t][r];
      if (ADD_OLD) v = (float)sO[(size_t)row * HIDDEN + col] + v;  // residual
      else         v = fmaxf(v, 0.f);                              // ReLU
      sO[(size_t)row * HIDDEN + col] = (bf16_t)v;
    }
  }
}

__global__ void k_mlp(const bf16_t* __restrict__ cb_bf,
                      const bf16_t* __restrict__ x_bf,
                      const float* __restrict__ residual,
                      const int* __restrict__ candIdx,
                      const bf16_t* __restrict__ wt_in,
                      const bf16_t* __restrict__ wt1,
                      const bf16_t* __restrict__ wt2,
                      const bf16_t* __restrict__ wt_out,
                      const float* __restrict__ b_in,
                      const float* __restrict__ b1,
                      const float* __restrict__ b2,
                      const float* __restrict__ b_out,
                      float* __restrict__ deltas,
                      float* __restrict__ distMat) {
  extern __shared__ char smem[];
  bf16_t* sX    = (bf16_t*)(smem);                 // 64 x 128
  bf16_t* sH    = (bf16_t*)(smem + 16384);         // 64 x 256
  bf16_t* sT    = (bf16_t*)(smem + 49152);         // 64 x 256
  bf16_t* sW    = (bf16_t*)(smem + 81920);         // up to 256 x 256
  int*    sCI   = (int*)  (smem + 212992);         // 64
  float*  sDist = (float*)(smem + 213248);         // 64

  const int tid  = threadIdx.x;
  const int wave = tid >> 5;
  const int lane = tid & 31;
  const int pairbase = blockIdx.x * 64;
  // LDS aperture maps generic addr[31:0] to the LDS byte offset.
  const uint32_t sW_off = (uint32_t)(uintptr_t)sW;

  if (tid < 64) { sCI[tid] = candIdx[pairbase + tid]; sDist[tid] = 0.f; }
  __syncthreads();

  // TDM: DMA W_in^T into LDS while all threads gather X = [code | x_prev].
  if (wave == 0) tdm_load_to_lds(wt_in, sW_off, (HIDDEN * 128 * 2) / 8);
  for (int i = tid; i < 64 * 128; i += 256) {
    int pl = i >> 7, d = i & 127;
    int rowB = (pairbase + pl) >> 6;
    sX[i] = (d < 64) ? cb_bf[(size_t)sCI[pl] * E_DIM + d]
                     : x_bf[(size_t)rowB * E_DIM + (d - 64)];
  }
  if (wave == 0) __builtin_amdgcn_s_wait_tensorcnt(0);
  __syncthreads();

  // H = relu(X @ W_in + b_in)
  mlp_gemm<128, false>(sX, sW, b_in, sH, wave, lane);
  __syncthreads();

  // 3 residual blocks (weights DMA'd by the TDM, one descriptor per matrix)
#pragma unroll 1
  for (int l = 0; l < NLAYERS; ++l) {
    if (wave == 0) {
      tdm_load_to_lds(wt1 + (size_t)l * HIDDEN * HIDDEN, sW_off,
                      (HIDDEN * HIDDEN * 2) / 8);
      __builtin_amdgcn_s_wait_tensorcnt(0);
    }
    __syncthreads();
    mlp_gemm<HIDDEN, false>(sH, sW, b1 + l * HIDDEN, sT, wave, lane);
    __syncthreads();
    if (wave == 0) {
      tdm_load_to_lds(wt2 + (size_t)l * HIDDEN * HIDDEN, sW_off,
                      (HIDDEN * HIDDEN * 2) / 8);
      __builtin_amdgcn_s_wait_tensorcnt(0);
    }
    __syncthreads();
    mlp_gemm<HIDDEN, true>(sT, sW, b2 + l * HIDDEN, sH, wave, lane);
    __syncthreads();
  }

  // out layer: delta = code + H @ W_out + b_out ; dist = ||residual - delta||^2
  if (wave == 0) {
    tdm_load_to_lds(wt_out, sW_off, (E_DIM * HIDDEN * 2) / 8);
    __builtin_amdgcn_s_wait_tensorcnt(0);
  }
  __syncthreads();
  {
    const int mt  = wave & 3;
    const int ntb = (wave >> 2) * 2;     // 2 column tiles (N = 64)
    const int n = lane & 15, hi = lane >> 4;
    v8f acc[2];
#pragma unroll
    for (int t = 0; t < 2; ++t) {
      float bv = b_out[(ntb + t) * 16 + n];
#pragma unroll
      for (int r = 0; r < 8; ++r) acc[t][r] = bv;
    }
#pragma unroll
    for (int ks = 0; ks < HIDDEN; ks += 32) {
      v16bf a = load_a_frag(sH, HIDDEN, mt * 16, ks, lane);
#pragma unroll
      for (int t = 0; t < 2; ++t) {
        v16bf b = load_b_frag(sW, HIDDEN, (ntb + t) * 16, ks, lane);
        acc[t] = wmma_bf16(a, b, acc[t]);
      }
    }
#pragma unroll
    for (int t = 0; t < 2; ++t) {
      int col = (ntb + t) * 16 + n;
#pragma unroll
      for (int r = 0; r < 8; ++r) {
        int pl   = mt * 16 + 8 * hi + r;
        int pair = pairbase + pl;
        float delta = acc[t][r] + (float)sX[(size_t)pl * 128 + col];
        deltas[(size_t)pair * E_DIM + col] = delta;
        int rowB = pair >> 6;
        float diff = residual[(size_t)rowB * E_DIM + col] - delta;
        atomicAdd(&sDist[pl], diff * diff);
      }
    }
  }
  __syncthreads();
  if (tid < 64) distMat[pairbase + tid] = sDist[tid];
}

// ---------------- phase 3: per-row argmin, z_q gather, loss/counts ----------

__global__ void k_pick(const float* __restrict__ distMat,
                       const int* __restrict__ candIdx,
                       const float* __restrict__ deltas,
                       const float* __restrict__ residual,
                       float* __restrict__ zq_out,
                       float* __restrict__ idx_out,
                       float* __restrict__ counts,
                       float* __restrict__ lossAcc) {
  __shared__ float sd[64];
  __shared__ int   si[64];
  __shared__ int   bestA;
  const int row = blockIdx.x, t = threadIdx.x;

  sd[t] = distMat[(size_t)row * TOP_A + t];
  si[t] = t;
  __syncthreads();
#pragma unroll
  for (int s = 32; s > 0; s >>= 1) {
    if (t < s && sd[t + s] < sd[t]) { sd[t] = sd[t + s]; si[t] = si[t + s]; }
    __syncthreads();
  }
  if (t == 0) bestA = si[0];
  __syncthreads();

  const int a = bestA;
  float zq = deltas[((size_t)row * TOP_A + a) * E_DIM + t];
  zq_out[(size_t)row * E_DIM + t] = zq;
  float d = zq - residual[(size_t)row * E_DIM + t];
  sd[t] = d * d;
  __syncthreads();
#pragma unroll
  for (int s = 32; s > 0; s >>= 1) {
    if (t < s) sd[t] += sd[t + s];
    __syncthreads();
  }
  if (t == 0) {
    atomicAdd(lossAcc, sd[0]);
    int ci = candIdx[(size_t)row * TOP_A + a];
    idx_out[row] = (float)ci;
    atomicAdd(&counts[ci], 1.f);
  }
}

__global__ void k_final(const float* __restrict__ counts,
                        const float* __restrict__ lossAcc,
                        float* __restrict__ out_loss,
                        float* __restrict__ out_perp) {
  __shared__ float s[256];
  float acc = 0.f;
  for (int i = threadIdx.x; i < N_E; i += 256) {
    float p = counts[i] * (1.f / (float)BATCH);
    acc += p * __logf(p + 1e-10f);
  }
  s[threadIdx.x] = acc;
  __syncthreads();
#pragma unroll
  for (int st = 128; st > 0; st >>= 1) {
    if (threadIdx.x < st) s[threadIdx.x] += s[threadIdx.x + st];
    __syncthreads();
  }
  if (threadIdx.x == 0) {
    *out_perp = __expf(-s[0]);
    *out_loss = (*lossAcc) * (1.25f / (float)(BATCH * E_DIM)); // embed + 0.25*commit
  }
}

// ---------------------------------------------------------------------------

extern "C" void kernel_launch(void* const* d_in, const int* in_sizes, int n_in,
                              void* d_out, int out_size, void* d_ws, size_t ws_size,
                              hipStream_t stream) {
  const float* residual = (const float*)d_in[0];   // 4096 x 64
  const float* x_prev   = (const float*)d_in[1];   // 4096 x 64
  const float* codebook = (const float*)d_in[2];   // 65536 x 64
  const float* W_in  = (const float*)d_in[3];      // 128 x 256
  const float* b_in  = (const float*)d_in[4];      // 256
  const float* W1    = (const float*)d_in[5];      // 3 x 256 x 256
  const float* b1    = (const float*)d_in[6];      // 3 x 256
  const float* W2    = (const float*)d_in[7];      // 3 x 256 x 256
  const float* b2    = (const float*)d_in[8];      // 3 x 256
  const float* W_out = (const float*)d_in[9];      // 256 x 64
  const float* b_out = (const float*)d_in[10];     // 64

  // workspace carve (all offsets 256B aligned)
  char* ws = (char*)d_ws;
  size_t off = 0;
  auto take = [&](size_t bytes) { char* p = ws + off; off = (off + bytes + 255) & ~(size_t)255; return p; };
  bf16_t* cb_bf   = (bf16_t*)take((size_t)N_E * E_DIM * 2);        // 8 MB
  bf16_t* res_bf  = (bf16_t*)take((size_t)BATCH * E_DIM * 2);
  bf16_t* x_bf    = (bf16_t*)take((size_t)BATCH * E_DIM * 2);
  bf16_t* wt_in   = (bf16_t*)take((size_t)HIDDEN * 128 * 2);
  bf16_t* wt1     = (bf16_t*)take((size_t)NLAYERS * HIDDEN * HIDDEN * 2);
  bf16_t* wt2     = (bf16_t*)take((size_t)NLAYERS * HIDDEN * HIDDEN * 2);
  bf16_t* wt_out  = (bf16_t*)take((size_t)E_DIM * HIDDEN * 2);
  float*  c2      = (float*)take((size_t)N_E * 4);
  float*  r2      = (float*)take((size_t)BATCH * 4);
  int*    candIdx = (int*)  take((size_t)NPAIRS * 4);
  float*  distMat = (float*)take((size_t)NPAIRS * 4);
  float*  deltas  = (float*)take((size_t)NPAIRS * E_DIM * 4);      // 64 MB
  float*  counts  = (float*)take((size_t)N_E * 4);
  float*  lossAcc = (float*)take(256);

  float* zq_out   = (float*)d_out;                       // 4096*64
  float* loss_out = zq_out + (size_t)BATCH * E_DIM;      // +262144
  float* perp_out = loss_out + 1;
  float* idx_out  = perp_out + 1;                        // 4096 floats

  // --- prep: bf16 conversions, transposed weights, squared norms ---
  k_cvt<<<(N_E * E_DIM + 255) / 256, 256, 0, stream>>>(codebook, cb_bf, N_E * E_DIM);
  k_cvt<<<(BATCH * E_DIM + 255) / 256, 256, 0, stream>>>(residual, res_bf, BATCH * E_DIM);
  k_cvt<<<(BATCH * E_DIM + 255) / 256, 256, 0, stream>>>(x_prev, x_bf, BATCH * E_DIM);
  k_cvt_t<<<(128 * HIDDEN + 255) / 256, 256, 0, stream>>>(W_in, wt_in, 128, HIDDEN);
  for (int l = 0; l < NLAYERS; ++l) {
    k_cvt_t<<<(HIDDEN * HIDDEN + 255) / 256, 256, 0, stream>>>(
        W1 + (size_t)l * HIDDEN * HIDDEN, wt1 + (size_t)l * HIDDEN * HIDDEN, HIDDEN, HIDDEN);
    k_cvt_t<<<(HIDDEN * HIDDEN + 255) / 256, 256, 0, stream>>>(
        W2 + (size_t)l * HIDDEN * HIDDEN, wt2 + (size_t)l * HIDDEN * HIDDEN, HIDDEN, HIDDEN);
  }
  k_cvt_t<<<(HIDDEN * E_DIM + 255) / 256, 256, 0, stream>>>(W_out, wt_out, HIDDEN, E_DIM);
  k_sqnorm<<<(N_E + 255) / 256, 256, 0, stream>>>(codebook, c2, N_E);
  k_sqnorm<<<(BATCH + 255) / 256, 256, 0, stream>>>(residual, r2, BATCH);

  // --- phase 1: distances + top-64 candidates ---
  k_dist_topk<<<BATCH / 16, 128, 0, stream>>>(res_bf, cb_bf, r2, c2, candIdx);

  // --- phase 2: MLP refinement of 262144 candidates ---
  k_mlp<<<NPAIRS / 64, 256, 213504, stream>>>(
      cb_bf, x_bf, residual, candIdx, wt_in, wt1, wt2, wt_out,
      b_in, b1, b2, b_out, deltas, distMat);

  // --- phase 3: select, outputs, loss, perplexity ---
  k_zero<<<(N_E + 255) / 256, 256, 0, stream>>>(counts, lossAcc);
  k_pick<<<BATCH, 64, 0, stream>>>(distMat, candIdx, deltas, residual,
                                   zq_out, idx_out, counts, lossAcc);
  k_final<<<1, 256, 0, stream>>>(counts, lossAcc, loss_out, perp_out);
}
